// MSMF_3384434229422
// MI455X (gfx1250) — matrix operand order repaired and checked by reading, not compiled, not run on hardware
//
#include <hip/hip_runtime.h>
#include <hip/hip_bf16.h>

// ---------------------------------------------------------------------------
// MSMF fused pipeline for MI455X (gfx1250, wave32, WMMA bf16 path)
//   Q = qw*F_b + qb ; K = kw*F_a + kb ; V = vw*F_a + vb     (WMMA GEMM)
//   s = K^T Q ; attn = softmax_m(s) ; M = V attn            (WMMA, 2-pass flash)
//   fused = F_b + M
//   out1 = relu(bn1(conv3x3(fused)))                        (WMMA, 9-tap GEMM)
//   out  = relu(bn2(conv3x3(out1)) + fused) + fused
// ---------------------------------------------------------------------------

typedef __attribute__((ext_vector_type(16))) __bf16 bf16x16;
typedef __attribute__((ext_vector_type(8)))  __bf16 bf16x8;
typedef __attribute__((ext_vector_type(4)))  __bf16 bf16x4;
typedef __attribute__((ext_vector_type(2)))  __bf16 bf16x2;
typedef __attribute__((ext_vector_type(8)))  float  f32x8;
typedef __attribute__((ext_vector_type(4)))  float  f32x4;

#define CD   256
#define HH   48
#define WW   48
#define NPIX (HH * WW)   // 2304
#define BB   16
#define KS   40          // padded LDS row stride in bf16 elements (80B: 16B-aligned, bank-clean)

__device__ __forceinline__ f32x8 wmma_bf16(bf16x16 a, bf16x16 b, f32x8 c) {
  // v_wmma_f32_16x16x32_bf16  D = A(16x32) * B(32x16) + C
  return __builtin_amdgcn_wmma_f32_16x16x32_bf16(false, a, false, b, (short)0, c,
                                                 false, false);
}

// A-operand: lane<16 holds row (lane&15), K elems 0..15 contiguous;
//            lane>=16 holds same row set, K elems 16..31.
// p must point at rowBase + (lane>>4)*16 ; rows have contiguous K.
__device__ __forceinline__ bf16x16 load_fragA(const __bf16* p) {
  union { bf16x16 v; bf16x8 h[2]; } u;
  u.h[0] = *(const bf16x8*)(p);
  u.h[1] = *(const bf16x8*)(p + 8);
  return u.v;
}

// B-operand: lane<16 holds column (lane&15), K elems 0..15 contiguous;
//            lane>=16 holds K elems 16..31.
// p must point at colBase + (lane>>4)*16 ; columns have contiguous K.
__device__ __forceinline__ bf16x16 load_fragB(const __bf16* p) {
  union { bf16x16 v; bf16x8 h[2]; } u;
  u.h[0] = *(const bf16x8*)(p);
  u.h[1] = *(const bf16x8*)(p + 8);
  return u.v;
}

// ---------------------------------------------------------------------------
// 1x1-conv projection:  D(o,n) = sum_c W(o,c) X(b,c,n) + bias(o)
// block = 256 thr (8 waves), tile 128(o) x 128(n), K-depth 32 per step.
// transposed==0 -> out (B,C,N) bf16 ; transposed==1 -> out (B,N,C) bf16
// ---------------------------------------------------------------------------
__global__ __launch_bounds__(256) void proj_kernel(
    const float* __restrict__ X, const float* __restrict__ W,
    const float* __restrict__ bias, __bf16* __restrict__ out, int transposed) {
  __shared__ __bf16 Al[128 * KS];
  __shared__ __bf16 Bl[128 * KS];
  const int b = blockIdx.z;
  const int n0 = blockIdx.x * 128;
  const int m0 = blockIdx.y * 128;
  const int tid = threadIdx.x, lane = tid & 31, wid = tid >> 5;
  const int wm = (wid & 1) * 64;   // wave's o-offset in block tile
  const int wn = (wid >> 1) * 32;  // wave's n-offset in block tile
  const int half = lane >> 4, row = lane & 15;
  const float* Xb = X + (size_t)b * CD * NPIX;

  f32x8 acc[4][2] = {};
  for (int k0 = 0; k0 < CD; k0 += 32) {
    // stage A (weights, row-major contiguous c): f32x4 load -> packed b64 store
    for (int i = 0; i < 4; ++i) {
      int idx = tid + i * 256;            // 0..1023 -> 128x8 quads
      int m = idx >> 3, k = (idx & 7) * 4;
      f32x4 w4 = *(const f32x4*)(W + (size_t)(m0 + m) * CD + k0 + k);
      bf16x4 p;
      p.x = (__bf16)w4.x; p.y = (__bf16)w4.y; p.z = (__bf16)w4.z; p.w = (__bf16)w4.w;
      *(bf16x4*)(&Al[m * KS + k]) = p;    // 8B-aligned (k mult of 4, rows 80B)
    }
    // stage B (activations, transpose (c,n)->[n][k]): 2 c-planes per thread,
    // coalesced in n, packed b32 store along k
    for (int i = 0; i < 8; ++i) {
      int k = (tid >> 7) * 2 + i * 4;     // pair base, covers k=0..31
      int n = tid & 127;
      const float* src = &Xb[(size_t)(k0 + k) * NPIX + n0 + n];
      bf16x2 p;
      p.x = (__bf16)src[0];
      p.y = (__bf16)src[NPIX];
      *(bf16x2*)(&Bl[n * KS + k]) = p;    // 4B-aligned
    }
    if (k0 + 32 < CD) {  // L2/L0 prefetch of next K tile (global_prefetch_b8)
      __builtin_prefetch(W + (size_t)(m0 + (tid >> 1)) * CD + k0 + 32, 0, 3);
      __builtin_prefetch(&Xb[(size_t)(k0 + 32 + (tid >> 7) * 16) * NPIX + n0 + (tid & 127)], 0, 3);
    }
    __syncthreads();
    bf16x16 af[4], bfr[2];
    for (int t = 0; t < 4; ++t)
      af[t] = load_fragA(&Al[(wm + t * 16 + row) * KS + half * 16]);
    for (int t = 0; t < 2; ++t)
      bfr[t] = load_fragB(&Bl[(wn + t * 16 + row) * KS + half * 16]);
    for (int tm = 0; tm < 4; ++tm)
      for (int tn = 0; tn < 2; ++tn)
        acc[tm][tn] = wmma_bf16(af[tm], bfr[tn], acc[tm][tn]);
    __syncthreads();
  }
  for (int tm = 0; tm < 4; ++tm)
    for (int tn = 0; tn < 2; ++tn)
      for (int r = 0; r < 8; ++r) {
        int o = m0 + wm + tm * 16 + r + half * 8;
        int n = n0 + wn + tn * 16 + row;
        float v = acc[tm][tn][r] + bias[o];
        if (!transposed) out[((size_t)b * CD + o) * NPIX + n] = (__bf16)v;
        else             out[((size_t)b * NPIX + n) * CD + o] = (__bf16)v;
      }
}

// ---------------------------------------------------------------------------
// Pass 1 of attention: per (b,n) running max & softmax denominator over m.
// s(n,m) = sum_c Kt(n,c) Qt(m,c); Kt is the A-operand, Qt the B-operand,
// both contraction-contiguous in global (L2-resident). Each wave owns 16 n.
// ---------------------------------------------------------------------------
__global__ __launch_bounds__(256) void stats_kernel(
    const __bf16* __restrict__ Kt, const __bf16* __restrict__ Qt,
    float2* __restrict__ stats) {
  const int b = blockIdx.y;
  const int tid = threadIdx.x, lane = tid & 31, wid = tid >> 5;
  const int n0 = blockIdx.x * 128 + wid * 16;
  const int half = lane >> 4, row = lane & 15;
  const __bf16* Ktb = Kt + ((size_t)b * NPIX + n0) * CD;
  const __bf16* Qtb = Qt + (size_t)b * NPIX * CD;

  bf16x16 af[8];
  for (int ks = 0; ks < 8; ++ks)
    af[ks] = load_fragA(&Ktb[(size_t)row * CD + ks * 32 + half * 16]);

  float run_mx[8], run_den[8];
  for (int r = 0; r < 8; ++r) { run_mx[r] = -3.0e38f; run_den[r] = 0.0f; }

  for (int m0 = 0; m0 < NPIX; m0 += 16) {
    if (m0 + 16 < NPIX)
      __builtin_prefetch(&Qtb[(size_t)(m0 + 16 + row) * CD + half * 128], 0, 3);
    f32x8 s = {};
    for (int ks = 0; ks < 8; ++ks) {
      bf16x16 bfr = load_fragB(&Qtb[(size_t)(m0 + row) * CD + ks * 32 + half * 16]);
      s = wmma_bf16(af[ks], bfr, s);
    }
    // D layout: VGPR r = row n0+r (lanes 0-15) / n0+8+r (lanes 16-31); col = lane&15
    for (int r = 0; r < 8; ++r) {
      float v = s[r];
      float mx = v;
      for (int off = 8; off; off >>= 1) mx = fmaxf(mx, __shfl_xor(mx, off));
      float nm = fmaxf(run_mx[r], mx);
      float e = __expf(v - nm);
      for (int off = 8; off; off >>= 1) e += __shfl_xor(e, off);
      run_den[r] = run_den[r] * __expf(run_mx[r] - nm) + e;
      run_mx[r] = nm;
    }
  }
  if (row == 0)  // lane 0 writes rows n0..n0+7, lane 16 writes n0+8..n0+15
    for (int r = 0; r < 8; ++r) {
      int n = n0 + r + half * 8;
      stats[(size_t)b * NPIX + n] = make_float2(run_mx[r], 1.0f / run_den[r]);
    }
}

// ---------------------------------------------------------------------------
// Pass 2: fused(b,c,m) = F_b + sum_n V(c,n) * exp(s(n,m)-mx(n))*invden(n)
// Score tiles recomputed by WMMA; the normalized P tile is converted
// in-register from two stacked D tiles into a B-operand (one cross-half
// shuffle), then consumed by the V-apply WMMA. Wave tile: 64(c) x 16(m).
// ---------------------------------------------------------------------------
__global__ __launch_bounds__(256) void attn_kernel(
    const __bf16* __restrict__ Kt, const __bf16* __restrict__ Qt,
    const __bf16* __restrict__ V, const float2* __restrict__ stats,
    const float* __restrict__ Fb, float* __restrict__ fused) {
  __shared__ float2 sst[NPIX];  // 18KB: all softmax stats for this batch
  const int b = blockIdx.y;
  const int tid = threadIdx.x, lane = tid & 31, wid = tid >> 5;
  for (int i = tid; i < NPIX; i += 256) sst[i] = stats[(size_t)b * NPIX + i];
  __syncthreads();

  const int m0 = blockIdx.x * 32 + (wid >> 2) * 16;
  const int c0 = (wid & 3) * 64;
  const int half = lane >> 4, row = lane & 15;
  const __bf16* Ktb = Kt + (size_t)b * NPIX * CD;
  const __bf16* Qtb = Qt + (size_t)b * NPIX * CD;
  const __bf16* Vb  = V  + (size_t)b * CD * NPIX;

  bf16x16 qf[8];
  for (int ks = 0; ks < 8; ++ks)
    qf[ks] = load_fragB(&Qtb[(size_t)(m0 + row) * CD + ks * 32 + half * 16]);

  f32x8 acc[4] = {};
  for (int nn = 0; nn < NPIX; nn += 32) {
    if (nn + 32 < NPIX) {  // prefetch next n-tile of K^T and V
      __builtin_prefetch(&Ktb[(size_t)(nn + 32 + row) * CD + half * 128], 0, 3);
      __builtin_prefetch(&Vb[(size_t)(c0 + lane) * NPIX + nn + 32], 0, 3);
    }
    f32x8 s0 = {}, s1 = {};
    for (int ks = 0; ks < 8; ++ks) {
      bf16x16 k0f = load_fragA(&Ktb[(size_t)(nn + row) * CD + ks * 32 + half * 16]);
      bf16x16 k1f = load_fragA(&Ktb[(size_t)(nn + 16 + row) * CD + ks * 32 + half * 16]);
      s0 = wmma_bf16(k0f, qf[ks], s0);
      s1 = wmma_bf16(k1f, qf[ks], s1);
    }
    // Build P as B-operand (K = n-rel 0..31, col = m).
    bf16x16 pf;
    for (int r = 0; r < 8; ++r) {
      float2 st0 = sst[nn + r + 8 * half];
      float2 st1 = sst[nn + 16 + r + 8 * half];
      float p0 = __expf(s0[r] - st0.x) * st0.y;  // own s0 row
      float p1 = __expf(s1[r] - st1.x) * st1.y;  // own s1 row
      float q0 = __shfl_xor(p0, 16);             // other half's s0 row
      float q1 = __shfl_xor(p1, 16);             // other half's s1 row
      // lane<16 needs K r (own s0) and K 8+r (peer s0);
      // lane>=16 needs K 16+r (peer s1) and K 24+r (own s1).
      pf[r]     = (__bf16)(half ? q1 : p0);
      pf[8 + r] = (__bf16)(half ? p1 : q0);
    }
    for (int t = 0; t < 4; ++t) {
      bf16x16 vf = load_fragA(&Vb[(size_t)(c0 + t * 16 + row) * NPIX + nn + half * 16]);
      acc[t] = wmma_bf16(vf, pf, acc[t]);
    }
  }
  for (int t = 0; t < 4; ++t)
    for (int r = 0; r < 8; ++r) {
      int c = c0 + t * 16 + r + half * 8;
      size_t idx = ((size_t)b * CD + c) * NPIX + m0 + row;
      fused[idx] = Fb[idx] + acc[t][r];
    }
}

// ---------------------------------------------------------------------------
// Reorder conv weights (O,C,3,3) f32 -> (tap,O,C) bf16 so the conv A-operand
// is contraction-contiguous and loads straight from L2.
// ---------------------------------------------------------------------------
__global__ void convw_kernel(const float* __restrict__ w, __bf16* __restrict__ out) {
  int idx = blockIdx.x * 256 + threadIdx.x;
  if (idx >= CD * CD * 9) return;
  int tap = idx % 9;
  int c = (idx / 9) % CD;
  int o = idx / (9 * CD);
  out[((size_t)tap * CD + o) * CD + c] = (__bf16)w[idx];
}

// ---------------------------------------------------------------------------
// conv3x3 + BN (+relu / +residual-relu-add), as 9 shifted GEMMs.
// Block: 256(o) x 96(px = 2 image rows); 8 waves = 4 o-groups x 2 px-groups;
// wave tile 64(o) x 48(px) = 4x3 WMMA accumulators. K loop: 9 taps x 8 c-steps.
// ---------------------------------------------------------------------------
__global__ __launch_bounds__(256) void conv_kernel(
    const float* __restrict__ in, const __bf16* __restrict__ Wc,
    const float* __restrict__ bn_s, const float* __restrict__ bn_b,
    const float* __restrict__ bn_m, const float* __restrict__ bn_v,
    const float* __restrict__ resid, float* __restrict__ out, int final_mode) {
  __shared__ __bf16 Bl[96 * KS];
  const int b = blockIdx.y;
  const int h0 = blockIdx.x * 2;
  const int tid = threadIdx.x, lane = tid & 31, wid = tid >> 5;
  const int o0 = (wid & 3) * 64;
  const int n0 = (wid >> 2) * 48;
  const int half = lane >> 4, row = lane & 15;
  const float* inb = in + (size_t)b * CD * NPIX;

  f32x8 acc[4][3] = {};
  for (int tap = 0; tap < 9; ++tap) {
    int dh = tap / 3 - 1, dw = tap % 3 - 1;
    for (int kc = 0; kc < CD; kc += 32) {
      __syncthreads();
      // stage shifted activations with halo zeroing: 96 px x 32 c,
      // 2 c-planes per thread packed into one b32 LDS store
      for (int i = 0; i < 6; ++i) {
        int idx = tid + i * 256;          // 0..1535 pairs
        int n = idx % 96, k = (idx / 96) * 2;
        int h = h0 + (n / 48) + dh, w = (n % 48) + dw;
        bf16x2 p; p.x = (__bf16)0.0f; p.y = (__bf16)0.0f;
        if (h >= 0 && h < HH && w >= 0 && w < WW) {
          const float* src = &inb[(size_t)(kc + k) * NPIX + h * WW + w];
          p.x = (__bf16)src[0];
          p.y = (__bf16)src[NPIX];
        }
        *(bf16x2*)(&Bl[n * KS + k]) = p;  // 4B-aligned
      }
      if (kc + 32 < CD)  // prefetch next c-slab of activations
        __builtin_prefetch(&inb[(size_t)(kc + 32 + (tid >> 5)) * NPIX + h0 * WW + lane * 3], 0, 3);
      __syncthreads();
      bf16x16 af[4];
      for (int t = 0; t < 4; ++t)
        af[t] = load_fragA(&Wc[((size_t)tap * CD + o0 + t * 16 + row) * CD + kc + half * 16]);
      for (int j = 0; j < 3; ++j) {
        bf16x16 bfr = load_fragB(&Bl[(n0 + j * 16 + row) * KS + half * 16]);
        for (int t = 0; t < 4; ++t)
          acc[t][j] = wmma_bf16(af[t], bfr, acc[t][j]);
      }
    }
  }
  for (int t = 0; t < 4; ++t)
    for (int r = 0; r < 8; ++r) {
      int o = o0 + t * 16 + r + half * 8;
      float inv = bn_s[o] * rsqrtf(bn_v[o] + 1e-5f);
      float beta = bn_b[o] - bn_m[o] * inv;
      for (int j = 0; j < 3; ++j) {
        float y = acc[t][j][r] * inv + beta;
        size_t idx = ((size_t)b * CD + o) * NPIX + h0 * WW + n0 + j * 16 + row;
        if (!final_mode) {
          out[idx] = fmaxf(y, 0.0f);
        } else {
          float fz = resid[idx];
          out[idx] = fmaxf(y + fz, 0.0f) + fz;
        }
      }
    }
}

// ---------------------------------------------------------------------------
extern "C" void kernel_launch(void* const* d_in, const int* in_sizes, int n_in,
                              void* d_out, int out_size, void* d_ws, size_t ws_size,
                              hipStream_t stream) {
  const float* F_b = (const float*)d_in[0];
  const float* F_a = (const float*)d_in[1];
  const float* qw = (const float*)d_in[2];
  const float* qb = (const float*)d_in[3];
  const float* kw = (const float*)d_in[4];
  const float* kb = (const float*)d_in[5];
  const float* vw = (const float*)d_in[6];
  const float* vb = (const float*)d_in[7];
  const float* c1w = (const float*)d_in[8];
  const float* bn1_s = (const float*)d_in[9];
  const float* bn1_b = (const float*)d_in[10];
  const float* bn1_m = (const float*)d_in[11];
  const float* bn1_v = (const float*)d_in[12];
  const float* c2w = (const float*)d_in[13];
  const float* bn2_s = (const float*)d_in[14];
  const float* bn2_b = (const float*)d_in[15];
  const float* bn2_m = (const float*)d_in[16];
  const float* bn2_v = (const float*)d_in[17];
  float* out = (float*)d_out;

  char* ws = (char*)d_ws;
  size_t off = 0;
  auto take = [&](size_t bytes) -> char* {
    char* p = ws + off;
    off += (bytes + 255) & ~(size_t)255;
    return p;
  };
  // ~134 MB of scratch
  __bf16* Qt = (__bf16*)take((size_t)BB * NPIX * CD * 2);  // (B,N,C)
  __bf16* Kt = (__bf16*)take((size_t)BB * NPIX * CD * 2);  // (B,N,C)
  __bf16* Vv = (__bf16*)take((size_t)BB * CD * NPIX * 2);  // (B,C,N)
  float2* stats = (float2*)take((size_t)BB * NPIX * sizeof(float2));
  float* fused = (float*)take((size_t)BB * CD * NPIX * 4);
  float* out1  = (float*)take((size_t)BB * CD * NPIX * 4);
  __bf16* Wc1 = (__bf16*)take((size_t)9 * CD * CD * 2);
  __bf16* Wc2 = (__bf16*)take((size_t)9 * CD * CD * 2);

  dim3 blk(256);
  convw_kernel<<<(CD * CD * 9 + 255) / 256, blk, 0, stream>>>(c1w, Wc1);
  convw_kernel<<<(CD * CD * 9 + 255) / 256, blk, 0, stream>>>(c2w, Wc2);

  dim3 gproj(NPIX / 128, CD / 128, BB);  // (18, 2, 16)
  proj_kernel<<<gproj, blk, 0, stream>>>(F_b, qw, qb, Qt, 1);
  proj_kernel<<<gproj, blk, 0, stream>>>(F_a, kw, kb, Kt, 1);
  proj_kernel<<<gproj, blk, 0, stream>>>(F_a, vw, vb, Vv, 0);

  stats_kernel<<<dim3(NPIX / 128, BB), blk, 0, stream>>>(Kt, Qt, stats);
  attn_kernel<<<dim3(NPIX / 32, BB), blk, 0, stream>>>(Kt, Qt, Vv, stats, F_b, fused);

  conv_kernel<<<dim3(HH / 2, BB), blk, 0, stream>>>(
      fused, Wc1, bn1_s, bn1_b, bn1_m, bn1_v, nullptr, out1, 0);
  conv_kernel<<<dim3(HH / 2, BB), blk, 0, stream>>>(
      out1, Wc2, bn2_s, bn2_b, bn2_m, bn2_v, fused, out, 1);
}
